// JointCrossAttentionBlock_14035953123691
// MI455X (gfx1250) — compile-verified, matrix operand order checked by reading, not run on hardware
//
#include <hip/hip_runtime.h>
#include <hip/hip_bf16.h>

#define DEV __device__ __forceinline__

typedef __attribute__((ext_vector_type(16))) __bf16         v16bf;
typedef __attribute__((ext_vector_type(8)))  float          v8f;
typedef __attribute__((ext_vector_type(4)))  int            v4i;
typedef __attribute__((ext_vector_type(8)))  unsigned short v8u16;
typedef __attribute__((ext_vector_type(16))) unsigned short v16u16;
typedef __attribute__((ext_vector_type(4)))  unsigned short v4u16;

static constexpr int BB = 2, NN = 2048, MM = 2048, DD = 1024;
static constexpr int H = 8, DH = 64, INNER = 512, FFD = 4096;
static constexpr int ROWS = BB * NN;   // 4096 token rows per side

// -------------------------------------------------- async LDS staging path ---
#if defined(__gfx1250__) && __has_builtin(__builtin_amdgcn_global_load_async_to_lds_b128)
#define ASYNC_LDS 1
#else
#define ASYNC_LDS 0
#endif

#if ASYNC_LDS
typedef __attribute__((address_space(1))) v4i glob_v4i;   // global int4
typedef __attribute__((address_space(3))) v4i lds_v4i;    // LDS int4
#endif

DEV void stage_b128(unsigned short* ldst, const unsigned short* gsrc) {
#if ASYNC_LDS
  __builtin_amdgcn_global_load_async_to_lds_b128(
      (glob_v4i*)gsrc, (lds_v4i*)ldst, 0, 0);
#else
  *(v8u16*)ldst = *(const v8u16*)gsrc;
#endif
}

DEV void wait_async0() {
#if ASYNC_LDS
#if __has_builtin(__builtin_amdgcn_s_wait_asynccnt)
  __builtin_amdgcn_s_wait_asynccnt(0);
#else
  asm volatile("s_wait_asynccnt 0x0" ::: "memory");
#endif
#endif
}

DEV void wait_async4() {   // allow the 4 just-issued prefetch ops to stay in flight
#if ASYNC_LDS
#if __has_builtin(__builtin_amdgcn_s_wait_asynccnt)
  __builtin_amdgcn_s_wait_asynccnt(4);
#else
  asm volatile("s_wait_asynccnt 0x4" ::: "memory");
#endif
#endif
}

// ---------------------------------------------------------------- helpers ----
DEV unsigned short f2bf(float f) {
  unsigned u = __builtin_bit_cast(unsigned, f);
  unsigned r = u + 0x7FFFu + ((u >> 16) & 1u);   // round-to-nearest-even
  return (unsigned short)(r >> 16);
}

DEV v8f zero8() {
  v8f z;
#pragma unroll
  for (int i = 0; i < 8; ++i) z[i] = 0.f;
  return z;
}

DEV v16bf ones_frag() {          // bf16 1.0 broadcast (B operand of row-sum WMMA)
  v16u16 t;
#pragma unroll
  for (int i = 0; i < 16; ++i) t[i] = 0x3F80u;
  return __builtin_bit_cast(v16bf, t);
}

// xor-swizzle max within 16-lane halves via ds_swizzle (group-of-32 xor mode)
template <int M>
DEV float swz_xor_max(float x) {
  int y = __builtin_amdgcn_ds_swizzle(__builtin_bit_cast(int, x), (M << 10) | 0x1f);
  return fmaxf(x, __builtin_bit_cast(float, y));
}

// Load one 16x32 bf16 fragment (A-pattern; assumed symmetric for B operand).
DEV v16bf load_frag(const unsigned short* base, int pitch) {
  const int lane = threadIdx.x & 31;
  const int r = lane & 15, kh = lane >> 4;
  const unsigned short* p = base + r * pitch + kh * 8;
  v8u16 lo = *(const v8u16*)(p);
  v8u16 hi = *(const v8u16*)(p + 16);
  v16u16 t;
#pragma unroll
  for (int i = 0; i < 8; ++i) { t[i] = lo[i]; t[i + 8] = hi[i]; }
  return __builtin_bit_cast(v16bf, t);
}

DEV v8f wmma_bf16(v16bf a, v16bf b, v8f c) {
  return __builtin_amdgcn_wmma_f32_16x16x32_bf16(false, a, false, b,
                                                 (short)0, c, false, false);
}

// ------------------------------------------------- weight convert+transpose --
// W: R x C fp32 row-major  ->  Wt: C x R bf16 row-major
__global__ __launch_bounds__(256)
void wconv_kernel(const float* __restrict__ W, unsigned short* __restrict__ Wt,
                  int R, int C) {
  __shared__ float tile[32][33];
  const int c0 = blockIdx.x * 32, r0 = blockIdx.y * 32;
  const int tx = threadIdx.x & 31, ty = threadIdx.x >> 5;
#pragma unroll
  for (int i = 0; i < 32; i += 8)
    tile[ty + i][tx] = W[(size_t)(r0 + ty + i) * C + c0 + tx];
  __syncthreads();
#pragma unroll
  for (int i = 0; i < 32; i += 8)
    Wt[(size_t)(c0 + ty + i) * R + r0 + tx] = f2bf(tile[tx][ty + i]);
}

// -------------------------------------------------------------- layernorm ----
__global__ __launch_bounds__(256)
void ln_kernel(const float* __restrict__ X, const float* __restrict__ g,
               const float* __restrict__ bt, unsigned short* __restrict__ Y) {
  const int row = blockIdx.x, tid = threadIdx.x;
  const float4 xv = ((const float4*)(X + (size_t)row * DD))[tid];
  float s  = xv.x + xv.y + xv.z + xv.w;
  float s2 = xv.x * xv.x + xv.y * xv.y + xv.z * xv.z + xv.w * xv.w;
#pragma unroll
  for (int off = 16; off; off >>= 1) {
    s  += __shfl_xor(s,  off, 32);
    s2 += __shfl_xor(s2, off, 32);
  }
  __shared__ float rs[8], rq[8];
  if ((tid & 31) == 0) { rs[tid >> 5] = s; rq[tid >> 5] = s2; }
  __syncthreads();
  s = 0.f; s2 = 0.f;
#pragma unroll
  for (int w = 0; w < 8; ++w) { s += rs[w]; s2 += rq[w]; }
  const float mean = s * (1.f / DD);
  const float var  = s2 * (1.f / DD) - mean * mean;
  const float rstd = rsqrtf(var + 1e-5f);
  const float4 gv = ((const float4*)g)[tid];
  const float4 bv = ((const float4*)bt)[tid];
  v4u16 o;
  o[0] = f2bf((xv.x - mean) * rstd * gv.x + bv.x);
  o[1] = f2bf((xv.y - mean) * rstd * gv.y + bv.y);
  o[2] = f2bf((xv.z - mean) * rstd * gv.z + bv.z);
  o[3] = f2bf((xv.w - mean) * rstd * gv.w + bv.w);
  *(v4u16*)(Y + (size_t)row * DD + tid * 4) = o;
}

// ------------------------------------------------------------------ GEMM -----
// C(MxN) = A(MxK,bf16) * Bt(NxK,bf16)^T.  WG 256 = 8 waves, tile 128x128,
// wave tile 64x32 (8 WMMA accumulators).  Double-buffered LDS with async
// global->LDS prefetch of tile k+1 overlapping WMMA on tile k.
// EPI 0: store bf16.  EPI 1: fp32 = residual + C + bias.  EPI 2: bf16 gelu(C+bias).
template <int EPI>
__global__ __launch_bounds__(256)
void gemm_bf16_kernel(const unsigned short* __restrict__ A,
                      const unsigned short* __restrict__ Bt,
                      int M, int N, int K,
                      const float* __restrict__ bias,
                      const float* __restrict__ residual,
                      float* __restrict__ outF,
                      unsigned short* __restrict__ outB) {
  constexpr int BM = 128, BN = 128, BK = 32, PA = 40, PB = 40;
  __shared__ unsigned short sA[2][BM * PA];
  __shared__ unsigned short sB[2][BN * PB];
  const int tid = threadIdx.x;
  const int bm = blockIdx.y * BM, bn = blockIdx.x * BN;
  const int wave = tid >> 5, lane = tid & 31;
  const int wm = (wave & 1) * 64, wn = (wave >> 1) * 32;

  v8f acc[4][2];
#pragma unroll
  for (int i = 0; i < 4; ++i) { acc[i][0] = zero8(); acc[i][1] = zero8(); }

  const int rT = tid >> 1, cT = (tid & 1) * 16;   // 128x32 tile staging map

  auto issue_tile = [&](unsigned short* dA, unsigned short* dB, int k0) {
    stage_b128(dA + rT * PA + cT,     A  + (size_t)(bm + rT) * K + k0 + cT);
    stage_b128(dA + rT * PA + cT + 8, A  + (size_t)(bm + rT) * K + k0 + cT + 8);
    stage_b128(dB + rT * PB + cT,     Bt + (size_t)(bn + rT) * K + k0 + cT);
    stage_b128(dB + rT * PB + cT + 8, Bt + (size_t)(bn + rT) * K + k0 + cT + 8);
  };
  auto compute_tile = [&](const unsigned short* cA, const unsigned short* cB) {
    v16bf af[4], bfr[2];
#pragma unroll
    for (int i = 0; i < 4; ++i) af[i] = load_frag(cA + (wm + i * 16) * PA, PA);
#pragma unroll
    for (int j = 0; j < 2; ++j) bfr[j] = load_frag(cB + (wn + j * 16) * PB, PB);
#pragma unroll
    for (int i = 0; i < 4; ++i)
#pragma unroll
      for (int j = 0; j < 2; ++j)
        acc[i][j] = wmma_bf16(af[i], bfr[j], acc[i][j]);
  };

  const int KT = K / BK;                 // even for all call sites (K % 64 == 0)
  issue_tile(sA[0], sB[0], 0);
  for (int kt = 0; kt < KT; kt += 2) {
    // ---- even tile: prefetch kt+1 into buffer 1, compute buffer 0 ----
    issue_tile(sA[1], sB[1], (kt + 1) * BK);
    wait_async4();                        // tile kt landed; kt+1 stays in flight
    __syncthreads();
    compute_tile(sA[0], sB[0]);
    __syncthreads();                      // all waves done reading buffer 0
    // ---- odd tile: prefetch kt+2 into buffer 0, compute buffer 1 ----
    if (kt + 2 < KT) {
      issue_tile(sA[0], sB[0], (kt + 2) * BK);
      wait_async4();
    } else {
      wait_async0();
    }
    __syncthreads();
    compute_tile(sA[1], sB[1]);
    __syncthreads();
  }

  const int rb = bm + wm + ((lane >> 4) * 8);
  const int cb = bn + wn + (lane & 15);
#pragma unroll
  for (int i = 0; i < 4; ++i)
#pragma unroll
    for (int j = 0; j < 2; ++j)
#pragma unroll
      for (int v = 0; v < 8; ++v) {
        const int row = rb + i * 16 + v;
        const int col = cb + j * 16;
        const size_t idx = (size_t)row * N + col;
        const float val = acc[i][j][v];
        if (EPI == 0) {
          outB[idx] = f2bf(val);
        } else if (EPI == 1) {
          outF[idx] = residual[idx] + val + bias[col];
        } else {
          const float t = val + bias[col];
          outB[idx] = f2bf(0.5f * t * (1.0f + erff(t * 0.70710678118654752f)));
        }
      }
}

// ------------------------------------------------------------ flash attn -----
// WG = 128 thr = 4 waves; wave owns 16 query rows; key tiles of 32.
// Online softmax; P bounced through LDS; row-sums via WMMA against ones;
// row-max via ds_swizzle xor reduction.
__global__ __launch_bounds__(128)
void flash_kernel(const unsigned short* __restrict__ Q,
                  const unsigned short* __restrict__ Kb,
                  const unsigned short* __restrict__ Vb,
                  const unsigned char* __restrict__ kmask,
                  unsigned short* __restrict__ O,
                  int Lq, int Lk) {
  constexpr int PK = 72, PV = 40, PP = 40;
  __shared__ unsigned short sK[32 * PK];        // [j][dh]
  __shared__ unsigned short sV[DH * PV];        // [dh][j] (transposed V)
  __shared__ unsigned short sP[4][16 * PP];     // per-wave P tile [16][32]
  const int b = blockIdx.z, h = blockIdx.y;
  const int tid = threadIdx.x, wave = tid >> 5, lane = tid & 31;
  const int q0 = blockIdx.x * 64 + wave * 16;
  const size_t qbase = ((size_t)b * Lq + q0) * INNER + h * DH;
  const size_t kbase = ((size_t)b * Lk) * INNER + h * DH;
  const float scale = 0.125f;                   // DH^-0.5

  const v16bf qa0 = load_frag(Q + qbase,      INNER);
  const v16bf qa1 = load_frag(Q + qbase + 32, INNER);
  const v16bf oneb = ones_frag();

  v8f o[4];
#pragma unroll
  for (int f = 0; f < 4; ++f) o[f] = zero8();
  float mrow[8], lrow[8];
#pragma unroll
  for (int v = 0; v < 8; ++v) { mrow[v] = -1e30f; lrow[v] = 0.f; }

  const int rS = tid >> 2;                // 0..31 (key row in tile)
  const int cS = (tid & 3) * 16;          // dh column group

  for (int j0 = 0; j0 < Lk; j0 += 32) {
    __syncthreads();
    // stage K tile 32x64 (async when available)
    stage_b128(sK + rS * PK + cS,     Kb + kbase + (size_t)(j0 + rS) * INNER + cS);
    stage_b128(sK + rS * PK + cS + 8, Kb + kbase + (size_t)(j0 + rS) * INNER + cS + 8);
    { // stage V tile transposed: sV[dh][j]
      const unsigned short* s = Vb + kbase + (size_t)(j0 + rS) * INNER + cS;
      v8u16 r0 = *(const v8u16*)s, r1 = *(const v8u16*)(s + 8);
#pragma unroll
      for (int d = 0; d < 8; ++d) {
        sV[(cS + d) * PV + rS]     = r0[d];
        sV[(cS + 8 + d) * PV + rS] = r1[d];
      }
    }
    wait_async0();
    __syncthreads();

    // S = Q * K^T (16x32), contract dh=64 in two k-chunks
    v8f s0 = zero8(), s1 = zero8();
    {
      v16bf k00 = load_frag(sK + 0 * PK + 0,  PK);
      v16bf k01 = load_frag(sK + 0 * PK + 32, PK);
      v16bf k10 = load_frag(sK + 16 * PK + 0,  PK);
      v16bf k11 = load_frag(sK + 16 * PK + 32, PK);
      s0 = wmma_bf16(qa0, k00, s0);
      s0 = wmma_bf16(qa1, k01, s0);
      s1 = wmma_bf16(qa0, k10, s1);
      s1 = wmma_bf16(qa1, k11, s1);
    }

    const int jc = j0 + (lane & 15);
    const bool km0 = kmask[(size_t)b * Lk + jc] != 0;
    const bool km1 = kmask[(size_t)b * Lk + jc + 16] != 0;

    float rmax[8];
#pragma unroll
    for (int v = 0; v < 8; ++v) {
      const float a = km0 ? s0[v] * scale : -1e30f;
      const float c = km1 ? s1[v] * scale : -1e30f;
      s0[v] = a; s1[v] = c;
      rmax[v] = fmaxf(a, c);
    }
#pragma unroll
    for (int v = 0; v < 8; ++v) {
      rmax[v] = swz_xor_max<8>(rmax[v]);
      rmax[v] = swz_xor_max<4>(rmax[v]);
      rmax[v] = swz_xor_max<2>(rmax[v]);
      rmax[v] = swz_xor_max<1>(rmax[v]);
    }

    float alpha[8];
#pragma unroll
    for (int v = 0; v < 8; ++v) {
      const float mn = fmaxf(mrow[v], rmax[v]);
      alpha[v] = __expf(mrow[v] - mn);
      mrow[v] = mn;
      const float p0 = __expf(s0[v] - mn);
      const float p1 = __expf(s1[v] - mn);
      const int prow = v + ((lane >> 4) * 8);
      sP[wave][prow * PP + (lane & 15)]      = f2bf(p0);
      sP[wave][prow * PP + 16 + (lane & 15)] = f2bf(p1);
      lrow[v] *= alpha[v];
    }
#pragma unroll
    for (int f = 0; f < 4; ++f)
#pragma unroll
      for (int v = 0; v < 8; ++v) o[f][v] *= alpha[v];

    asm volatile("s_wait_dscnt 0" ::: "memory");   // own-wave P stores visible

    const v16bf pa = load_frag(sP[wave], PP);      // 16x32 A-frag of P
    const v8f rs = wmma_bf16(pa, oneb, zero8());   // row sums via matrix pipe
#pragma unroll
    for (int v = 0; v < 8; ++v) lrow[v] += rs[v];
#pragma unroll
    for (int f = 0; f < 4; ++f) {
      v16bf vf = load_frag(sV + (f * 16) * PV, PV);
      o[f] = wmma_bf16(pa, vf, o[f]);
    }
  }

#pragma unroll
  for (int f = 0; f < 4; ++f)
#pragma unroll
    for (int v = 0; v < 8; ++v) {
      const int row = q0 + v + ((lane >> 4) * 8);
      const int col = h * DH + f * 16 + (lane & 15);
      O[((size_t)b * Lq + row) * INNER + col] = f2bf(o[f][v] / lrow[v]);
    }
}

// ------------------------------------------------------------------ host -----
extern "C" void kernel_launch(void* const* d_in, const int* in_sizes, int n_in,
                              void* d_out, int out_size, void* d_ws, size_t ws_size,
                              hipStream_t stream) {
  (void)in_sizes; (void)n_in; (void)out_size; (void)ws_size;
  const float* x      = (const float*)d_in[0];
  const float* ctx    = (const float*)d_in[1];
  const unsigned char* mask  = (const unsigned char*)d_in[2];
  const unsigned char* cmask = (const unsigned char*)d_in[3];
  const float* ln_x_g = (const float*)d_in[4];
  const float* ln_x_b = (const float*)d_in[5];
  const float* ln_c_g = (const float*)d_in[6];
  const float* ln_c_b = (const float*)d_in[7];
  const float* W_qk   = (const float*)d_in[8];
  const float* W_v    = (const float*)d_in[9];
  const float* Wc_qk  = (const float*)d_in[10];
  const float* Wc_v   = (const float*)d_in[11];
  const float* W_out  = (const float*)d_in[12];
  const float* b_out  = (const float*)d_in[13];
  const float* Wc_out = (const float*)d_in[14];
  const float* bc_out = (const float*)d_in[15];
  const float* ff_g   = (const float*)d_in[16];
  const float* ff_b   = (const float*)d_in[17];
  const float* ff_w1  = (const float*)d_in[18];
  const float* ff_b1  = (const float*)d_in[19];
  const float* ff_w2  = (const float*)d_in[20];
  const float* ff_b2  = (const float*)d_in[21];
  const float* cff_g  = (const float*)d_in[22];
  const float* cff_b  = (const float*)d_in[23];
  const float* cff_w1 = (const float*)d_in[24];
  const float* cff_b1 = (const float*)d_in[25];
  const float* cff_w2 = (const float*)d_in[26];
  const float* cff_b2 = (const float*)d_in[27];

  float* out_x = (float*)d_out;
  float* out_c = out_x + (size_t)ROWS * DD;

  char* p = (char*)d_ws;
  auto take = [&](size_t bytes) { char* q = p; p += (bytes + 255) & ~(size_t)255; return q; };
  auto U16 = [&](size_t n) { return (unsigned short*)take(n * 2); };
  auto F32 = [&](size_t n) { return (float*)take(n * 4); };

  unsigned short* wt_qk  = U16((size_t)INNER * DD);
  unsigned short* wt_v   = U16((size_t)INNER * DD);
  unsigned short* wt_cqk = U16((size_t)INNER * DD);
  unsigned short* wt_cv  = U16((size_t)INNER * DD);
  unsigned short* wt_out = U16((size_t)DD * INNER);
  unsigned short* wt_cou = U16((size_t)DD * INNER);
  unsigned short* wt_fw1 = U16((size_t)FFD * DD);
  unsigned short* wt_fw2 = U16((size_t)DD * FFD);
  unsigned short* wt_cw1 = U16((size_t)FFD * DD);
  unsigned short* wt_cw2 = U16((size_t)DD * FFD);
  unsigned short* xn     = U16((size_t)ROWS * DD);
  unsigned short* cn     = U16((size_t)ROWS * DD);
  unsigned short* qkb    = U16((size_t)ROWS * INNER);
  unsigned short* vb     = U16((size_t)ROWS * INNER);
  unsigned short* cqkb   = U16((size_t)ROWS * INNER);
  unsigned short* cvb    = U16((size_t)ROWS * INNER);
  unsigned short* attob  = U16((size_t)ROWS * INNER);
  unsigned short* cattob = U16((size_t)ROWS * INNER);
  float*          x1     = F32((size_t)ROWS * DD);
  float*          c1     = F32((size_t)ROWS * DD);
  unsigned short* ln2    = U16((size_t)ROWS * DD);
  unsigned short* hbuf   = U16((size_t)ROWS * FFD);

  const dim3 tb(256);
  auto T = [&](const float* W, unsigned short* Wt, int R, int C) {
    wconv_kernel<<<dim3(C / 32, R / 32), tb, 0, stream>>>(W, Wt, R, C);
  };
  T(W_qk, wt_qk, DD, INNER);   T(W_v, wt_v, DD, INNER);
  T(Wc_qk, wt_cqk, DD, INNER); T(Wc_v, wt_cv, DD, INNER);
  T(W_out, wt_out, INNER, DD); T(Wc_out, wt_cou, INNER, DD);
  T(ff_w1, wt_fw1, DD, FFD);   T(ff_w2, wt_fw2, FFD, DD);
  T(cff_w1, wt_cw1, DD, FFD);  T(cff_w2, wt_cw2, FFD, DD);

  ln_kernel<<<ROWS, tb, 0, stream>>>(x,   ln_x_g, ln_x_b, xn);
  ln_kernel<<<ROWS, tb, 0, stream>>>(ctx, ln_c_g, ln_c_b, cn);

  auto G0 = [&](const unsigned short* A, const unsigned short* Bt,
                unsigned short* Ob, int M, int Nn, int K) {
    gemm_bf16_kernel<0><<<dim3(Nn / 128, M / 128), tb, 0, stream>>>(
        A, Bt, M, Nn, K, nullptr, nullptr, nullptr, Ob);
  };
  G0(xn, wt_qk,  qkb,  ROWS, INNER, DD);
  G0(xn, wt_v,   vb,   ROWS, INNER, DD);
  G0(cn, wt_cqk, cqkb, ROWS, INNER, DD);
  G0(cn, wt_cv,  cvb,  ROWS, INNER, DD);

  flash_kernel<<<dim3(NN / 64, H, BB), dim3(128), 0, stream>>>(
      qkb, cqkb, cvb, cmask, attob, NN, MM);
  flash_kernel<<<dim3(MM / 64, H, BB), dim3(128), 0, stream>>>(
      cqkb, qkb, vb, mask, cattob, MM, NN);

  gemm_bf16_kernel<1><<<dim3(DD / 128, ROWS / 128), tb, 0, stream>>>(
      attob, wt_out, ROWS, DD, INNER, b_out, x, x1, nullptr);
  gemm_bf16_kernel<1><<<dim3(DD / 128, ROWS / 128), tb, 0, stream>>>(
      cattob, wt_cou, ROWS, DD, INNER, bc_out, ctx, c1, nullptr);

  // x-side FFN
  ln_kernel<<<ROWS, tb, 0, stream>>>(x1, ff_g, ff_b, ln2);
  gemm_bf16_kernel<2><<<dim3(FFD / 128, ROWS / 128), tb, 0, stream>>>(
      ln2, wt_fw1, ROWS, FFD, DD, ff_b1, nullptr, nullptr, hbuf);
  gemm_bf16_kernel<1><<<dim3(DD / 128, ROWS / 128), tb, 0, stream>>>(
      hbuf, wt_fw2, ROWS, DD, FFD, ff_b2, x1, out_x, nullptr);

  // context-side FFN (reuses ln2/hbuf; stream order serializes)
  ln_kernel<<<ROWS, tb, 0, stream>>>(c1, cff_g, cff_b, ln2);
  gemm_bf16_kernel<2><<<dim3(FFD / 128, ROWS / 128), tb, 0, stream>>>(
      ln2, wt_cw1, ROWS, FFD, DD, cff_b1, nullptr, nullptr, hbuf);
  gemm_bf16_kernel<1><<<dim3(DD / 128, ROWS / 128), tb, 0, stream>>>(
      hbuf, wt_cw2, ROWS, DD, FFD, cff_b2, c1, out_c, nullptr);
}
